// MMDLoss_25374666785208
// MI455X (gfx1250) — matrix operand order repaired and checked by reading, not compiled, not run on hardware
//
#include <hip/hip_runtime.h>

typedef __attribute__((ext_vector_type(16))) __bf16 v16bf;
typedef __attribute__((ext_vector_type(8)))  __bf16 v8bf;
typedef __attribute__((ext_vector_type(8)))  float  v8f;
typedef __attribute__((ext_vector_type(4)))  float  f32x4;

#define NROWS 8192
#define DIM   256
#define BM    128          // block tile rows (M)
#define BN    64           // block tile cols (N)
#define GBX   (NROWS / BM) // 64
#define GBY   (NROWS / BN) // 128
#define NBLK  (GBX * GBY)  // 8192 partial sums per gram

// ===========================================================================
// FAST PATH: bf16 hi/lo planes precomputed in workspace; gram kernel loads
// WMMA fragments directly with global_load_b128 (no per-step conversion VALU).
// ===========================================================================

// ---------------------------------------------------------------------------
// Preprocess: fp32 row -> bf16 hi plane, bf16 lo plane (x = hi + lo), plus
// row squared-norm. One wave per row; blockIdx.y selects X or Y.
// ---------------------------------------------------------------------------
__global__ __launch_bounds__(256) void split_norm_kernel(
    const float* __restrict__ X, const float* __restrict__ Y,
    __bf16* __restrict__ Xhi, __bf16* __restrict__ Xlo,
    __bf16* __restrict__ Yhi, __bf16* __restrict__ Ylo,
    float* __restrict__ x2, float* __restrict__ y2) {
    const float*  src = blockIdx.y ? Y   : X;
    __bf16*       dhi = blockIdx.y ? Yhi : Xhi;
    __bf16*       dlo = blockIdx.y ? Ylo : Xlo;
    float*        dn  = blockIdx.y ? y2  : x2;
    const int tid  = threadIdx.x;
    const int wave = tid >> 5;
    const int lane = tid & 31;
    const int row  = blockIdx.x * 8 + wave;
    const float* p = src + row * DIM;
    float s = 0.0f;
#pragma unroll
    for (int j = 0; j < DIM / 32; ++j) {
        const int idx = lane + 32 * j;
        float v = p[idx];
        s += v * v;
        __bf16 h = (__bf16)v;
        dhi[row * DIM + idx] = h;
        dlo[row * DIM + idx] = (__bf16)(v - (float)h);
    }
#pragma unroll
    for (int off = 16; off > 0; off >>= 1) s += __shfl_xor(s, off, 32);
    if (lane == 0) dn[row] = s;
}

// ---------------------------------------------------------------------------
// Fragment loaders from bf16 planes (pure b128 loads + register concat).
// A-matrix 16x32 bf16 (ISA 7.12.2): lane L holds row M=L%16; elems 0..7 =
// K[koff..koff+7], elems 8..15 = K[koff+16..koff+23], koff = (L>=16)*8.
// ---------------------------------------------------------------------------
__device__ __forceinline__ v16bf load_a_bf(const __bf16* __restrict__ plane,
                                           int row, int kbase, int lane) {
    const int koff = (lane >> 4) * 8;
    const __bf16* p = plane + row * DIM + kbase + koff;
    v8bf a = *(const v8bf*)p;
    v8bf b = *(const v8bf*)(p + 16);
    return __builtin_shufflevector(a, b, 0, 1, 2, 3, 4, 5, 6, 7,
                                   8, 9, 10, 11, 12, 13, 14, 15);
}

// B-matrix 32x16 bf16: lane L holds column N=L%16; elems 0..7 =
// K[koff..koff+7], elems 8..15 = K[koff+16..koff+23], koff = (L>=16)*8.
__device__ __forceinline__ v16bf load_b_bf(const __bf16* __restrict__ plane,
                                           int col, int kbase, int lane) {
    const int koff = (lane >> 4) * 8;
    const __bf16* p = plane + col * DIM + kbase + koff;
    v8bf a = *(const v8bf*)p;
    v8bf b = *(const v8bf*)(p + 16);
    return __builtin_shufflevector(a, b, 0, 1, 2, 3, 4, 5, 6, 7,
                                   8, 9, 10, 11, 12, 13, 14, 15);
}

// ---------------------------------------------------------------------------
// Fused Gram + RBF + partial-sum, split-bf16 WMMA (hi*hi + hi*lo + lo*hi).
// Block tile 128x64; 8 waves arranged 4(M) x 2(N), each wave 32x32 (2x2 tiles).
// ---------------------------------------------------------------------------
__global__ __launch_bounds__(256) void gram_rbf_sum_bf16(
    const __bf16* __restrict__ Ahi, const __bf16* __restrict__ Alo,
    const __bf16* __restrict__ Bhi, const __bf16* __restrict__ Blo,
    const float* __restrict__ a2, const float* __restrict__ b2,
    const int* __restrict__ sigma_p, int symmetric,
    float* __restrict__ partial) {
    const int tid  = threadIdx.x;
    const int wave = tid >> 5;
    const int lane = tid & 31;
    const int blockM = blockIdx.x * BM;
    const int blockN = blockIdx.y * BN;
    const int blockId = blockIdx.y * gridDim.x + blockIdx.x;

    __shared__ float wave_sums[8];

    if (symmetric && (blockM >= blockN + BN)) {   // strictly lower triangle
        if (tid == 0) partial[blockId] = 0.0f;
        return;
    }

    const float inv_sigma = 1.0f / (float)(*sigma_p);
    const int waveM = blockM + (wave & 3) * 32;
    const int waveN = blockN + (wave >> 2) * 32;
    const int sub   = lane & 15;

    v8f acc[2][2] = {};

#pragma unroll 2
    for (int kb = 0; kb < DIM; kb += 32) {
        v16bf ahi[2], alo[2], bhi[2], blo[2];
#pragma unroll
        for (int t = 0; t < 2; ++t) {
            ahi[t] = load_a_bf(Ahi, waveM + t * 16 + sub, kb, lane);
            alo[t] = load_a_bf(Alo, waveM + t * 16 + sub, kb, lane);
            bhi[t] = load_b_bf(Bhi, waveN + t * 16 + sub, kb, lane);
            blo[t] = load_b_bf(Blo, waveN + t * 16 + sub, kb, lane);
        }
#pragma unroll
        for (int i = 0; i < 2; ++i)
#pragma unroll
            for (int j = 0; j < 2; ++j) {
                acc[i][j] = __builtin_amdgcn_wmma_f32_16x16x32_bf16(
                    false, ahi[i], false, bhi[j], (short)0, acc[i][j], false, false);
                acc[i][j] = __builtin_amdgcn_wmma_f32_16x16x32_bf16(
                    false, ahi[i], false, blo[j], (short)0, acc[i][j], false, false);
                acc[i][j] = __builtin_amdgcn_wmma_f32_16x16x32_bf16(
                    false, alo[i], false, bhi[j], (short)0, acc[i][j], false, false);
            }
    }

    // C/D layout: VGPR r, lanes 0-15 -> M=r, N=lane; lanes 16-31 -> M=r+8.
    const int mtop = (lane >> 4) * 8;
    float lsum = 0.0f;
#pragma unroll
    for (int i = 0; i < 2; ++i)
#pragma unroll
        for (int j = 0; j < 2; ++j) {
            const int n = waveN + j * 16 + sub;
            const float bn2 = b2[n];
#pragma unroll
            for (int r = 0; r < 8; ++r) {
                const int m = waveM + i * 16 + mtop + r;
                const float d = a2[m] + bn2 - 2.0f * acc[i][j][r];
                float e = __expf(-d * inv_sigma);
                if (symmetric)
                    e *= (m < n) ? 2.0f : ((m == n) ? 1.0f : 0.0f);
                lsum += e;
            }
        }

#pragma unroll
    for (int off = 16; off > 0; off >>= 1) lsum += __shfl_xor(lsum, off, 32);
    if (lane == 0) wave_sums[wave] = lsum;
    __syncthreads();
    if (tid == 0) {
        float s = 0.0f;
#pragma unroll
        for (int w = 0; w < 8; ++w) s += wave_sums[w];
        partial[blockId] = s;
    }
}

// ===========================================================================
// FALLBACK PATH (small workspace): on-the-fly fp32 -> split-bf16 conversion.
// ===========================================================================
__device__ __forceinline__ void load_a_f32(const float* __restrict__ base,
                                           int row, int kbase, int lane,
                                           v16bf& hi, v16bf& lo) {
    const int koff = (lane >> 4) * 8;
    const float* p = base + row * DIM + kbase + koff;
    f32x4 f0 = *(const f32x4*)(p);
    f32x4 f1 = *(const f32x4*)(p + 4);
    f32x4 f2 = *(const f32x4*)(p + 16);
    f32x4 f3 = *(const f32x4*)(p + 20);
    float v[16];
#pragma unroll
    for (int i = 0; i < 4; ++i) { v[i] = f0[i]; v[4+i] = f1[i]; v[8+i] = f2[i]; v[12+i] = f3[i]; }
#pragma unroll
    for (int i = 0; i < 16; ++i) {
        __bf16 h = (__bf16)v[i];
        hi[i] = h;
        lo[i] = (__bf16)(v[i] - (float)h);
    }
}

__device__ __forceinline__ void load_b_f32(const float* __restrict__ base,
                                           int col, int kbase, int lane,
                                           v16bf& hi, v16bf& lo) {
    const int koff = (lane >> 4) * 8;
    const float* p = base + col * DIM + kbase + koff;
    f32x4 f0 = *(const f32x4*)(p);
    f32x4 f1 = *(const f32x4*)(p + 4);
    f32x4 f2 = *(const f32x4*)(p + 16);
    f32x4 f3 = *(const f32x4*)(p + 20);
    float v[16];
#pragma unroll
    for (int i = 0; i < 4; ++i) { v[i] = f0[i]; v[4+i] = f1[i]; v[8+i] = f2[i]; v[12+i] = f3[i]; }
#pragma unroll
    for (int i = 0; i < 16; ++i) {
        __bf16 h = (__bf16)v[i];
        hi[i] = h;
        lo[i] = (__bf16)(v[i] - (float)h);
    }
}

__global__ __launch_bounds__(256) void rownorm_kernel(
    const float* __restrict__ X, const float* __restrict__ Y,
    float* __restrict__ x2, float* __restrict__ y2) {
    const float* src = blockIdx.y ? Y : X;
    float* dst = blockIdx.y ? y2 : x2;
    const int tid = threadIdx.x, wave = tid >> 5, lane = tid & 31;
    const int row = blockIdx.x * 8 + wave;
    const float* p = src + row * DIM;
    float s = 0.0f;
#pragma unroll
    for (int j = 0; j < DIM / 32; ++j) { float v = p[lane + 32 * j]; s += v * v; }
#pragma unroll
    for (int off = 16; off > 0; off >>= 1) s += __shfl_xor(s, off, 32);
    if (lane == 0) dst[row] = s;
}

__global__ __launch_bounds__(256) void gram_rbf_sum_f32(
    const float* __restrict__ A, const float* __restrict__ B,
    const float* __restrict__ a2, const float* __restrict__ b2,
    const int* __restrict__ sigma_p, int symmetric,
    float* __restrict__ partial) {
    const int tid  = threadIdx.x;
    const int wave = tid >> 5;
    const int lane = tid & 31;
    const int blockM = blockIdx.x * BM;
    const int blockN = blockIdx.y * BN;
    const int blockId = blockIdx.y * gridDim.x + blockIdx.x;

    __shared__ float wave_sums[8];

    if (symmetric && (blockM >= blockN + BN)) {
        if (tid == 0) partial[blockId] = 0.0f;
        return;
    }

    const float inv_sigma = 1.0f / (float)(*sigma_p);
    const int waveM = blockM + (wave & 3) * 32;
    const int waveN = blockN + (wave >> 2) * 32;
    const int sub   = lane & 15;

    v8f acc[2][2] = {};

    for (int kb = 0; kb < DIM; kb += 32) {
        v16bf ahi[2], alo[2], bhi[2], blo[2];
#pragma unroll
        for (int t = 0; t < 2; ++t) {
            load_a_f32(A, waveM + t * 16 + sub, kb, lane, ahi[t], alo[t]);
            load_b_f32(B, waveN + t * 16 + sub, kb, lane, bhi[t], blo[t]);
        }
#pragma unroll
        for (int i = 0; i < 2; ++i)
#pragma unroll
            for (int j = 0; j < 2; ++j) {
                acc[i][j] = __builtin_amdgcn_wmma_f32_16x16x32_bf16(
                    false, ahi[i], false, bhi[j], (short)0, acc[i][j], false, false);
                acc[i][j] = __builtin_amdgcn_wmma_f32_16x16x32_bf16(
                    false, ahi[i], false, blo[j], (short)0, acc[i][j], false, false);
                acc[i][j] = __builtin_amdgcn_wmma_f32_16x16x32_bf16(
                    false, alo[i], false, bhi[j], (short)0, acc[i][j], false, false);
            }
    }

    const int mtop = (lane >> 4) * 8;
    float lsum = 0.0f;
#pragma unroll
    for (int i = 0; i < 2; ++i)
#pragma unroll
        for (int j = 0; j < 2; ++j) {
            const int n = waveN + j * 16 + sub;
            const float bn2 = b2[n];
#pragma unroll
            for (int r = 0; r < 8; ++r) {
                const int m = waveM + i * 16 + mtop + r;
                const float d = a2[m] + bn2 - 2.0f * acc[i][j][r];
                float e = __expf(-d * inv_sigma);
                if (symmetric)
                    e *= (m < n) ? 2.0f : ((m == n) ? 1.0f : 0.0f);
                lsum += e;
            }
        }

#pragma unroll
    for (int off = 16; off > 0; off >>= 1) lsum += __shfl_xor(lsum, off, 32);
    if (lane == 0) wave_sums[wave] = lsum;
    __syncthreads();
    if (tid == 0) {
        float s = 0.0f;
#pragma unroll
        for (int w = 0; w < 8; ++w) s += wave_sums[w];
        partial[blockId] = s;
    }
}

// ---------------------------------------------------------------------------
// Single-block deterministic final reduction:
// out = mean(XX) + mean(YY) - 2*mean(XY)
// ---------------------------------------------------------------------------
__global__ __launch_bounds__(256) void final_reduce_kernel(
    const float* __restrict__ partials, float* __restrict__ out) {
    __shared__ double sh0[256], sh1[256], sh2[256];
    const int tid = threadIdx.x;
    double s0 = 0.0, s1 = 0.0, s2 = 0.0;
    for (int i = tid; i < NBLK; i += 256) {
        s0 += (double)partials[i];
        s1 += (double)partials[NBLK + i];
        s2 += (double)partials[2 * NBLK + i];
    }
    sh0[tid] = s0; sh1[tid] = s1; sh2[tid] = s2;
    __syncthreads();
    for (int off = 128; off > 0; off >>= 1) {
        if (tid < off) {
            sh0[tid] += sh0[tid + off];
            sh1[tid] += sh1[tid + off];
            sh2[tid] += sh2[tid + off];
        }
        __syncthreads();
    }
    if (tid == 0) {
        const double invNN = 1.0 / ((double)NROWS * (double)NROWS);
        out[0] = (float)((sh0[0] + sh1[0] - 2.0 * sh2[0]) * invNN);
    }
}

// ---------------------------------------------------------------------------
extern "C" void kernel_launch(void* const* d_in, const int* in_sizes, int n_in,
                              void* d_out, int out_size, void* d_ws, size_t ws_size,
                              hipStream_t stream) {
    (void)in_sizes; (void)n_in; (void)out_size;
    const float* X = (const float*)d_in[0];
    const float* Y = (const float*)d_in[1];
    const int* sigma = (const int*)d_in[2];
    float* out = (float*)d_out;

    const size_t PLANE = (size_t)NROWS * DIM;              // elements per plane
    const size_t BIG_NEED = 4 * PLANE * sizeof(__bf16)     // hi/lo planes
                          + 2 * NROWS * sizeof(float)      // norms
                          + 3 * NBLK * sizeof(float);      // partials

    dim3 grid(GBX, GBY);

    if (ws_size >= BIG_NEED) {
        // Fast path: precomputed bf16 hi/lo planes (L2-resident, ~16.9 MB).
        char* w = (char*)d_ws;
        __bf16* Xhi = (__bf16*)w;                 w += PLANE * sizeof(__bf16);
        __bf16* Xlo = (__bf16*)w;                 w += PLANE * sizeof(__bf16);
        __bf16* Yhi = (__bf16*)w;                 w += PLANE * sizeof(__bf16);
        __bf16* Ylo = (__bf16*)w;                 w += PLANE * sizeof(__bf16);
        float*  x2  = (float*)w;                  w += NROWS * sizeof(float);
        float*  y2  = (float*)w;                  w += NROWS * sizeof(float);
        float*  part = (float*)w;

        split_norm_kernel<<<dim3(NROWS / 8, 2), 256, 0, stream>>>(
            X, Y, Xhi, Xlo, Yhi, Ylo, x2, y2);

        gram_rbf_sum_bf16<<<grid, 256, 0, stream>>>(Xhi, Xlo, Xhi, Xlo, x2, x2,
                                                    sigma, 1, part);
        gram_rbf_sum_bf16<<<grid, 256, 0, stream>>>(Yhi, Ylo, Yhi, Ylo, y2, y2,
                                                    sigma, 1, part + NBLK);
        gram_rbf_sum_bf16<<<grid, 256, 0, stream>>>(Xhi, Xlo, Yhi, Ylo, x2, y2,
                                                    sigma, 0, part + 2 * NBLK);
        final_reduce_kernel<<<1, 256, 0, stream>>>(part, out);
    } else {
        // Fallback: on-the-fly conversion, tiny workspace (~160 KB).
        float* ws = (float*)d_ws;
        float* x2 = ws;
        float* y2 = ws + NROWS;
        float* part = ws + 2 * NROWS;

        rownorm_kernel<<<dim3(NROWS / 8, 2), 256, 0, stream>>>(X, Y, x2, y2);
        gram_rbf_sum_f32<<<grid, 256, 0, stream>>>(X, X, x2, x2, sigma, 1, part);
        gram_rbf_sum_f32<<<grid, 256, 0, stream>>>(Y, Y, y2, y2, sigma, 1, part + NBLK);
        gram_rbf_sum_f32<<<grid, 256, 0, stream>>>(X, Y, x2, y2, sigma, 0, part + 2 * NBLK);
        final_reduce_kernel<<<1, 256, 0, stream>>>(part, out);
    }
}
